// GNNFraudDetection_73521250173700
// MI455X (gfx1250) — matrix operand order repaired and checked by reading, not compiled, not run on hardware
//
#include <hip/hip_runtime.h>
#include <hip/hip_bf16.h>
#include <math.h>

typedef __attribute__((ext_vector_type(16))) _Float16 v16h;
typedef __attribute__((ext_vector_type(8)))  float    v8f;

#define H 64
#define NG 1024
#define NTYPES 6
#define BSTRIDE 72   // padded LDS row stride (halves): 144 B, 16B-aligned

// ---------------------------------------------------------------------------
// helpers
// ---------------------------------------------------------------------------
__device__ inline void atomicMaxFloat(float* addr, float val) {
  // standard monotone-bit-pattern trick; location must be initialized to -inf
  if (val >= 0.0f) atomicMax((int*)addr, __float_as_int(val));
  else             atomicMin((unsigned int*)addr, __float_as_uint(val));
}

__global__ void fill_kernel(float* p, float v, long n) {
  long i = (long)blockIdx.x * blockDim.x + threadIdx.x;
  if (i < n) p[i] = v;
}

// ---------------------------------------------------------------------------
// embedding: Y[n,h] = sum_k X[n,k] * W[k,h] + b[h]   (din is 5..12, tiny)
// ---------------------------------------------------------------------------
__global__ void embed_kernel(const float* __restrict__ X, const float* __restrict__ W,
                             const float* __restrict__ b, float* __restrict__ Y,
                             long N, int din) {
  long gid = (long)blockIdx.x * blockDim.x + threadIdx.x;
  if (gid >= N * H) return;
  long n = gid >> 6; int h = gid & 63;
  float acc = b[h];
  for (int k = 0; k < din; ++k) acc += X[n * din + k] * W[k * H + h];
  Y[gid] = acc;
}

// ---------------------------------------------------------------------------
// WMMA GEMM: Y = X @ W  (X: Nx64 f32, W: 64x64 f32, Y: Nx64 f32)
// flags: 1 = add bias, 2 = relu.
// Block = 128 threads = 4 waves. W is staged TRANSPOSED in LDS as f16 so each
// B fragment is one contiguous 32B (b128-aligned) LDS read. Each wave computes
// a 32-row x 64-col tile: 2 M-tiles x 4 N-tiles x 2 K-steps = 16 WMMAs, with
// each B fragment reused for both M-tiles.
// ---------------------------------------------------------------------------
__global__ __launch_bounds__(128)
void gemm64_wmma_kernel(const float* __restrict__ X, const float* __restrict__ Wmat,
                        const float* __restrict__ bias, float* __restrict__ Y,
                        long N, int flags) {
  __shared__ _Float16 Bt[H * BSTRIDE];             // W transposed: Bt[n*BSTRIDE + k]
  int t = threadIdx.x;
  for (int i = t; i < H * H; i += 128) {
    int k = i >> 6, n = i & 63;                    // coalesced global read of W[k][n]
    Bt[n * BSTRIDE + k] = (_Float16)Wmat[i];
  }
  __syncthreads();

  int wave = t >> 5;
  int lane = t & 31;
  int half = lane >> 4;     // half-wave: K split for A/B fragments
  int mloc = lane & 15;     // M within tile (A) / N within tile (B, D)

  long rowBase = (long)blockIdx.x * 128 + (long)wave * 32;   // 32 rows per wave
  const float* xr[2];
#pragma unroll
  for (int mt = 0; mt < 2; ++mt) {
    long row = rowBase + mt * 16 + mloc;
    long rowc = row < N ? row : (N - 1);           // clamp: keep EXEC all-ones
    xr[mt] = X + rowc * H;
  }

  v8f acc[2][4];
#pragma unroll
  for (int mt = 0; mt < 2; ++mt)
#pragma unroll
    for (int nt = 0; nt < 4; ++nt) acc[mt][nt] = (v8f)0.0f;

#pragma unroll
  for (int ks = 0; ks < 2; ++ks) {
    int k0 = ks * 32;
    // A fragments: per lane, two contiguous 8-float runs -> float4 loads + cvt
    v16h a[2];
#pragma unroll
    for (int mt = 0; mt < 2; ++mt) {
      const float4* p1 = (const float4*)(xr[mt] + k0 + half * 16);       // K = k0+half*16 .. +7
      const float4* p2 = (const float4*)(xr[mt] + k0 + half * 16 + 8);   // K = k0+half*16+8 .. +7
      float4 f0 = p1[0], f1 = p1[1], f2 = p2[0], f3 = p2[1];
      a[mt][0]  = (_Float16)f0.x; a[mt][1]  = (_Float16)f0.y;
      a[mt][2]  = (_Float16)f0.z; a[mt][3]  = (_Float16)f0.w;
      a[mt][4]  = (_Float16)f1.x; a[mt][5]  = (_Float16)f1.y;
      a[mt][6]  = (_Float16)f1.z; a[mt][7]  = (_Float16)f1.w;
      a[mt][8]  = (_Float16)f2.x; a[mt][9]  = (_Float16)f2.y;
      a[mt][10] = (_Float16)f2.z; a[mt][11] = (_Float16)f2.w;
      a[mt][12] = (_Float16)f3.x; a[mt][13] = (_Float16)f3.y;
      a[mt][14] = (_Float16)f3.z; a[mt][15] = (_Float16)f3.w;
    }
#pragma unroll
    for (int nt = 0; nt < 4; ++nt) {
      // B fragment: 16 contiguous halves in transposed LDS, 16B-aligned
      int ncol = nt * 16 + mloc;
      v16h b = *(const v16h*)&Bt[ncol * BSTRIDE + k0 + half * 16];
      acc[0][nt] = __builtin_amdgcn_wmma_f32_16x16x32_f16(
          false, a[0], false, b, (short)0, acc[0][nt], false, false);
      acc[1][nt] = __builtin_amdgcn_wmma_f32_16x16x32_f16(
          false, a[1], false, b, (short)0, acc[1][nt], false, false);
    }
  }

  // D layout: VGPR r -> M = 8*half + r, N = mloc (+16*nt)
#pragma unroll
  for (int mt = 0; mt < 2; ++mt) {
#pragma unroll
    for (int nt = 0; nt < 4; ++nt) {
#pragma unroll
      for (int r = 0; r < 8; ++r) {
        long orow = rowBase + mt * 16 + half * 8 + r;
        if (orow < N) {
          int ncol = nt * 16 + mloc;
          float v = acc[mt][nt][r];
          if (flags & 1) v += bias[ncol];
          if (flags & 2) v = v > 0.0f ? v : 0.0f;
          Y[orow * H + ncol] = v;
        }
      }
    }
  }
}

// ---------------------------------------------------------------------------
// wa[k] = sum_h W[k,h] * a[h]    (fold attention vector through the linear)
// ---------------------------------------------------------------------------
__global__ void matvec64_kernel(const float* __restrict__ W, const float* __restrict__ a,
                                float* __restrict__ wa) {
  int k = threadIdx.x;   // 64 threads
  float acc = 0.0f;
  for (int h = 0; h < H; ++h) acc += W[k * H + h] * a[h];
  wa[k] = acc;
}

// out[n] = dot(X[n,:], wa)
__global__ void rowdot_kernel(const float* __restrict__ X, const float* __restrict__ wa,
                              float* __restrict__ out, long N) {
  long n = (long)blockIdx.x * blockDim.x + threadIdx.x;
  if (n >= N) return;
  const float4* xr = (const float4*)(X + n * H);
  float acc = 0.0f;
#pragma unroll
  for (int i = 0; i < 16; ++i) {
    float4 v = xr[i];
    acc += v.x * wa[4 * i] + v.y * wa[4 * i + 1] + v.z * wa[4 * i + 2] + v.w * wa[4 * i + 3];
  }
  out[n] = acc;
}

// ---------------------------------------------------------------------------
// edge passes
// ---------------------------------------------------------------------------
__global__ void edge_max_kernel(const int* __restrict__ src, const int* __restrict__ dst,
                                const float* __restrict__ as, const float* __restrict__ ad,
                                float* __restrict__ m, long E) {
  long e = (long)blockIdx.x * blockDim.x + threadIdx.x;
  if (e >= E) return;
  __builtin_prefetch(src + e + 8192, 0, 1);        // global_prefetch_b8
  __builtin_prefetch(dst + e + 8192, 0, 1);
  float v = as[src[e]] + ad[dst[e]];
  v = v > 0.0f ? v : 0.2f * v;                     // leaky_relu(0.2)
  atomicMaxFloat(&m[dst[e]], v);
}

__global__ void edge_exp_kernel(const int* __restrict__ src, const int* __restrict__ dst,
                                const float* __restrict__ as, const float* __restrict__ ad,
                                const float* __restrict__ m, float* __restrict__ p,
                                float* __restrict__ s, long E) {
  long e = (long)blockIdx.x * blockDim.x + threadIdx.x;
  if (e >= E) return;
  __builtin_prefetch(src + e + 8192, 0, 1);
  __builtin_prefetch(dst + e + 8192, 0, 1);
  int d = dst[e];
  float v = as[src[e]] + ad[d];
  v = v > 0.0f ? v : 0.2f * v;
  float pe = __expf(v - m[d]);
  p[e] = pe;
  atomicAdd(&s[d], pe);
}

// tmp[dst, h] += p[e] * xs[src, h]
__global__ void edge_scatter_kernel(const int* __restrict__ src, const int* __restrict__ dst,
                                    const float* __restrict__ p, const float* __restrict__ xs,
                                    float* __restrict__ tmp, long E) {
  long gid = (long)blockIdx.x * blockDim.x + threadIdx.x;
  if (gid >= E * H) return;
  long e = gid >> 6; int h = gid & 63;
  atomicAdd(&tmp[(long)dst[e] * H + h], p[e] * xs[(long)src[e] * H + h]);
}

// agg[n,h] += tmp[n,h] / (s[n] + 1e-16) + b[h]
__global__ void finalize_rel_kernel(const float* __restrict__ tmp, const float* __restrict__ s,
                                    const float* __restrict__ b, float* __restrict__ agg, long N) {
  long gid = (long)blockIdx.x * blockDim.x + threadIdx.x;
  if (gid >= N * H) return;
  long n = gid >> 6; int h = gid & 63;
  agg[gid] += tmp[gid] / (s[n] + 1e-16f) + b[h];
}

__global__ void relu_copy_kernel(const float* __restrict__ a, float* __restrict__ x, long n) {
  long i = (long)blockIdx.x * blockDim.x + threadIdx.x;
  if (i < n) { float v = a[i]; x[i] = v > 0.0f ? v : 0.0f; }
}

// ---------------------------------------------------------------------------
// pooling + classifier
// ---------------------------------------------------------------------------
__global__ void pool_accum_kernel(const float* __restrict__ tx, const int* __restrict__ batch,
                                  float* __restrict__ meanb, float* __restrict__ maxb,
                                  float* __restrict__ cnt, long NTX) {
  long gid = (long)blockIdx.x * blockDim.x + threadIdx.x;
  if (gid >= NTX * H) return;
  long n = gid >> 6; int h = gid & 63;
  int g = batch[n];
  float v = tx[gid];
  atomicAdd(&meanb[(long)g * H + h], v);
  atomicMaxFloat(&maxb[(long)g * H + h], v);
  if (h == 0) atomicAdd(&cnt[g], 1.0f);
}

__global__ void pool_out_kernel(const float* __restrict__ meanb, const float* __restrict__ maxb,
                                const float* __restrict__ cnt, const float* __restrict__ pW,
                                const float* __restrict__ pb, float* __restrict__ pooled) {
  int gid = blockIdx.x * blockDim.x + threadIdx.x;   // NG*H = 65536
  if (gid >= NG * H) return;
  int g = gid >> 6; int h = gid & 63;
  float c = cnt[g]; c = c > 1.0f ? c : 1.0f;
  float acc = pb[h];
  for (int k = 0; k < H; ++k) {
    float mn = meanb[g * H + k] / c;
    float mx = maxb[g * H + k];
    mx = (mx > -3.0e38f) ? mx : 0.0f;                // -inf from empty group -> 0
    acc += mn * pW[k * H + h] + mx * pW[(H + k) * H + h];
  }
  pooled[gid] = acc;
}

__global__ void cls_out_kernel(const float* __restrict__ hbuf, const float* __restrict__ W2,
                               const float* __restrict__ b2, float* __restrict__ out, long N) {
  long n = (long)blockIdx.x * blockDim.x + threadIdx.x;
  if (n >= N) return;
  const float4* hr = (const float4*)(hbuf + n * H);
  float acc = b2[0];
#pragma unroll
  for (int i = 0; i < 16; ++i) {
    float4 v = hr[i];
    acc += v.x * W2[4 * i] + v.y * W2[4 * i + 1] + v.z * W2[4 * i + 2] + v.w * W2[4 * i + 3];
  }
  out[n] = 1.0f / (1.0f + __expf(-acc));
}

// ---------------------------------------------------------------------------
// host orchestration
// ---------------------------------------------------------------------------
extern "C" void kernel_launch(void* const* d_in, const int* in_sizes, int n_in,
                              void* d_out, int out_size, void* d_ws, size_t ws_size,
                              hipStream_t stream) {
  (void)n_in; (void)out_size; (void)ws_size;
  static const long N_NODES[NTYPES] = {100000, 150000, 500000, 20000, 5000, 50000};
  static const int  FEAT[NTYPES]    = {8, 6, 12, 6, 5, 5};
  static const int  REL_S[12] = {0, 1, 2, 2, 2, 2, 1, 2, 3, 4, 5, 2};
  static const int  REL_D[12] = {1, 2, 3, 4, 5, 2, 0, 1, 2, 2, 2, 2};
  const long NMAX = 500000, NTX = 500000;
  const int  L = 3;

  // -------- inputs --------
  const float* x_in[NTYPES]; const float* embW[NTYPES]; const float* embB[NTYPES];
  for (int t = 0; t < NTYPES; ++t) {
    x_in[t] = (const float*)d_in[t];
    embW[t] = (const float*)d_in[6 + 2 * t];
    embB[t] = (const float*)d_in[7 + 2 * t];
  }
  const float* gat_W  = (const float*)d_in[18];   // [3,12,64,64]
  const float* gat_as = (const float*)d_in[19];   // [3,12,64]
  const float* gat_ad = (const float*)d_in[20];
  const float* gat_b  = (const float*)d_in[21];
  const float* pool_W = (const float*)d_in[22];   // [128,64]
  const float* pool_b = (const float*)d_in[23];
  const float* cls_W1 = (const float*)d_in[24];
  const float* cls_b1 = (const float*)d_in[25];
  const float* cls_W2 = (const float*)d_in[26];
  const float* cls_b2 = (const float*)d_in[27];
  const int* esrc[12]; const int* edst[12]; long ecnt[12];
  for (int r = 0; r < 12; ++r) {
    esrc[r] = (const int*)d_in[28 + 2 * r];
    edst[r] = (const int*)d_in[29 + 2 * r];
    ecnt[r] = (long)in_sizes[28 + 2 * r];
  }
  const int* batch = (const int*)d_in[52];

  float* out_scores = (float*)d_out;
  float* out_pooled = out_scores + NTX;

  // -------- workspace layout --------
  char* ws = (char*)d_ws;
  size_t off = 0;
  auto alloc = [&](size_t bytes) -> size_t {
    size_t o = off; off += (bytes + 255) & ~(size_t)255; return o;
  };
  float* xb[NTYPES]; float* ab[NTYPES];
  for (int t = 0; t < NTYPES; ++t) xb[t] = (float*)(ws + alloc(N_NODES[t] * H * 4));
  for (int t = 0; t < NTYPES; ++t) ab[t] = (float*)(ws + alloc(N_NODES[t] * H * 4));
  float* xs_buf  = (float*)(ws + alloc(NMAX * H * 4));
  float* tmp_buf = (float*)(ws + alloc(NMAX * H * 4));
  float* as_buf  = (float*)(ws + alloc(NMAX * 4));
  float* ad_buf  = (float*)(ws + alloc(NMAX * 4));
  float* m_buf   = (float*)(ws + alloc(NMAX * 4));
  float* s_buf   = (float*)(ws + alloc(NMAX * 4));
  float* p_buf   = (float*)(ws + alloc(NMAX * 4));
  float* wa_s    = (float*)(ws + alloc(H * 4));
  float* wa_d    = (float*)(ws + alloc(H * 4));
  float* cnt_buf = (float*)(ws + alloc(NG * 4));
  float* mean_buf= (float*)(ws + alloc(NG * H * 4));
  float* max_buf = (float*)(ws + alloc(NG * H * 4));

  const int TPB = 256;
  auto nblk = [](long n, int tpb) -> unsigned { return (unsigned)((n + tpb - 1) / tpb); };

  // -------- embeddings --------
  for (int t = 0; t < NTYPES; ++t)
    embed_kernel<<<nblk(N_NODES[t] * H, TPB), TPB, 0, stream>>>(
        x_in[t], embW[t], embB[t], xb[t], N_NODES[t], FEAT[t]);

  // -------- GAT layers --------
  for (int l = 0; l < L; ++l) {
    for (int t = 0; t < NTYPES; ++t)
      hipMemsetAsync(ab[t], 0, N_NODES[t] * H * 4, stream);

    for (int r = 0; r < 12; ++r) {
      int st = REL_S[r], dt = REL_D[r];
      long Ns = N_NODES[st], Nd = N_NODES[dt], E = ecnt[r];
      const float* Wlr = gat_W  + ((size_t)(l * 12 + r)) * H * H;
      const float* aS  = gat_as + ((size_t)(l * 12 + r)) * H;
      const float* aD  = gat_ad + ((size_t)(l * 12 + r)) * H;
      const float* bLr = gat_b  + ((size_t)(l * 12 + r)) * H;

      // fold attention vectors through W, then per-node scalar logits
      matvec64_kernel<<<1, H, 0, stream>>>(Wlr, aS, wa_s);
      matvec64_kernel<<<1, H, 0, stream>>>(Wlr, aD, wa_d);
      rowdot_kernel<<<nblk(Ns, TPB), TPB, 0, stream>>>(xb[st], wa_s, as_buf, Ns);
      rowdot_kernel<<<nblk(Nd, TPB), TPB, 0, stream>>>(xb[dt], wa_d, ad_buf, Nd);

      // xs = x_src @ W   (WMMA)
      gemm64_wmma_kernel<<<nblk(Ns, 128), 128, 0, stream>>>(
          xb[st], Wlr, nullptr, xs_buf, Ns, 0);

      // segment softmax over dst
      fill_kernel<<<nblk(Nd, TPB), TPB, 0, stream>>>(m_buf, -INFINITY, Nd);
      hipMemsetAsync(s_buf, 0, Nd * 4, stream);
      hipMemsetAsync(tmp_buf, 0, Nd * H * 4, stream);
      edge_max_kernel<<<nblk(E, TPB), TPB, 0, stream>>>(esrc[r], edst[r], as_buf, ad_buf, m_buf, E);
      edge_exp_kernel<<<nblk(E, TPB), TPB, 0, stream>>>(esrc[r], edst[r], as_buf, ad_buf, m_buf, p_buf, s_buf, E);
      edge_scatter_kernel<<<nblk(E * H, TPB), TPB, 0, stream>>>(esrc[r], edst[r], p_buf, xs_buf, tmp_buf, E);
      finalize_rel_kernel<<<nblk(Nd * H, TPB), TPB, 0, stream>>>(tmp_buf, s_buf, bLr, ab[dt], Nd);
    }

    for (int t = 0; t < NTYPES; ++t)
      relu_copy_kernel<<<nblk(N_NODES[t] * H, TPB), TPB, 0, stream>>>(ab[t], xb[t], N_NODES[t] * H);
  }

  // -------- pooling --------
  hipMemsetAsync(cnt_buf, 0, NG * 4, stream);
  hipMemsetAsync(mean_buf, 0, NG * H * 4, stream);
  fill_kernel<<<nblk((long)NG * H, TPB), TPB, 0, stream>>>(max_buf, -INFINITY, (long)NG * H);
  pool_accum_kernel<<<nblk(NTX * H, TPB), TPB, 0, stream>>>(xb[2], batch, mean_buf, max_buf, cnt_buf, NTX);
  pool_out_kernel<<<nblk((long)NG * H, TPB), TPB, 0, stream>>>(mean_buf, max_buf, cnt_buf, pool_W, pool_b, out_pooled);

  // -------- classifier --------
  // h = relu(tx @ W1 + b1)  (WMMA, bias+relu fused), reuse xs_buf
  gemm64_wmma_kernel<<<nblk(NTX, 128), 128, 0, stream>>>(xb[2], cls_W1, cls_b1, xs_buf, NTX, 3);
  cls_out_kernel<<<nblk(NTX, TPB), TPB, 0, stream>>>(xs_buf, cls_W2, cls_b2, out_scores, NTX);
}
